// GraphFourierTransform_76184129897200
// MI455X (gfx1250) — matrix operand order repaired and checked by reading, not compiled, not run on hardware
//
#include <hip/hip_runtime.h>
#include <hip/hip_bf16.h>

// Problem constants (match reference)
#define BB   8
#define NN   2048
#define CINC 128
#define COUTC 256
#define KNN  16

typedef __attribute__((ext_vector_type(16))) __bf16 v16bf;
typedef __attribute__((ext_vector_type(8)))  float  v8f;

__device__ __forceinline__ unsigned short f2bf(float f) {
  unsigned int u = __float_as_uint(f);
  unsigned int r = u + 0x7FFFu + ((u >> 16) & 1u);          // round-to-nearest-even
  if ((u & 0x7F800000u) == 0x7F800000u) r = u;              // inf/nan: truncate
  return (unsigned short)(r >> 16);
}

// ---------------------------------------------------------------------------
// Prep: fold bias+BN into (alpha,beta), pack W_spec / [W_low;W_high] into the
// per-lane bf16 A order used here with V_WMMA_F32_16X16X32_BF16 (contiguous K, B matches):
// lane l (row M=l%16, half=l/16), element i -> K = 16*half + i.
// ---------------------------------------------------------------------------
__global__ __launch_bounds__(256) void prep_kernel(
    const float* __restrict__ W_spec, const float* __restrict__ b_spec,
    const float* __restrict__ gamma,  const float* __restrict__ beta,
    const float* __restrict__ mean,   const float* __restrict__ var,
    const float* __restrict__ W_low,  const float* __restrict__ b_low,
    const float* __restrict__ W_high, const float* __restrict__ b_high,
    unsigned short* __restrict__ ApackS, unsigned short* __restrict__ ApackLH,
    float* __restrict__ alpha, float* __restrict__ betaC, float* __restrict__ bias2) {
  const int tid = threadIdx.x;
  {
    const int o = tid;                                       // 256 threads = COUT
    const float a = gamma[o] * rsqrtf(var[o] + 1e-5f);
    alpha[o] = a;
    betaC[o] = (b_spec[o] - mean[o]) * a + beta[o];
    bias2[o] = (o < 128) ? b_low[o] : b_high[o - 128];
  }
  // W_spec: 16 oTiles x 4 kTiles x 32 lanes x 16 vals
  for (int e = tid; e < 16 * 4 * 32 * 16; e += 256) {
    const int i = e & 15, lane = (e >> 4) & 31, kT = (e >> 9) & 3, oT = e >> 11;
    const int half = lane >> 4, v = i >> 1, p = i & 1;
    const int kk = 2 * v + p + 16 * half;
    const int o = oT * 16 + (lane & 15), k = kT * 32 + kk;
    ApackS[e] = f2bf(W_spec[o * CINC + k]);
  }
  // combined [W_low;W_high]: 16 oTiles x 8 kTiles x 32 lanes x 16 vals
  for (int e = tid; e < 16 * 8 * 32 * 16; e += 256) {
    const int i = e & 15, lane = (e >> 4) & 31, kT = (e >> 9) & 7, oT = e >> 12;
    const int half = lane >> 4, v = i >> 1, p = i & 1;
    const int kk = 2 * v + p + 16 * half;
    const int o = oT * 16 + (lane & 15), k = kT * 32 + kk;
    const float val = (o < 128) ? W_low[o * COUTC + k] : W_high[(o - 128) * COUTC + k];
    ApackLH[e] = f2bf(val);
  }
}

// ---------------------------------------------------------------------------
// Feature transpose: (B,C,N) f32 -> (B,N,C) f32, classic 32x32 LDS tile.
// ---------------------------------------------------------------------------
__global__ __launch_bounds__(256) void transpose_kernel(
    const float* __restrict__ feat, float* __restrict__ featT) {
  __shared__ float tile[32][33];
  const int b = blockIdx.z, n0 = blockIdx.x * 32, c0 = blockIdx.y * 32;
  const int tx = threadIdx.x, ty = threadIdx.y;            // (32,8)
#pragma unroll
  for (int i = 0; i < 4; ++i) {
    const int c = c0 + ty + i * 8;
    tile[ty + i * 8][tx] = feat[((size_t)b * CINC + c) * NN + n0 + tx];
  }
  __syncthreads();
#pragma unroll
  for (int i = 0; i < 4; ++i) {
    const int n = n0 + ty + i * 8;
    featT[((size_t)b * NN + n) * CINC + c0 + tx] = tile[tx][ty + i * 8];
  }
}

// ---------------------------------------------------------------------------
// Brute-force kNN (K=16 smallest of sq[n]+sq[m]-2*dot), xyz tiled via LDS.
// One thread per query point; per-thread sorted insertion list (early-out).
// Also counts in-degrees with atomics (reverse CSR step 1).
// ---------------------------------------------------------------------------
__global__ __launch_bounds__(256) void knn_kernel(
    const float* __restrict__ xyz, int* __restrict__ idx, int* __restrict__ indeg) {
  const int b = blockIdx.x >> 3;
  const int n = ((blockIdx.x & 7) << 8) + threadIdx.x;
  const float* xb = xyz + (size_t)b * 3 * NN;
  const float px = xb[n], py = xb[NN + n], pz = xb[2 * NN + n];
  const float psq = px * px + py * py + pz * pz;
  float bestD[KNN]; int bestI[KNN];
#pragma unroll
  for (int k = 0; k < KNN; ++k) { bestD[k] = 3.4e38f; bestI[k] = 0; }
  __shared__ float sx[256], sy[256], sz[256], ssq[256];
  for (int t0 = 0; t0 < NN; t0 += 256) {
    __syncthreads();
    const float x = xb[t0 + threadIdx.x];
    const float y = xb[NN + t0 + threadIdx.x];
    const float z = xb[2 * NN + t0 + threadIdx.x];
    sx[threadIdx.x] = x; sy[threadIdx.x] = y; sz[threadIdx.x] = z;
    ssq[threadIdx.x] = x * x + y * y + z * z;
    __syncthreads();
    for (int j = 0; j < 256; ++j) {
      float d = psq + ssq[j] - 2.0f * (px * sx[j] + py * sy[j] + pz * sz[j]);
      if (d < bestD[KNN - 1]) {
        int m = t0 + j;
#pragma unroll
        for (int k = 0; k < KNN; ++k) {
          if (d < bestD[k]) {
            float td = bestD[k]; int ti = bestI[k];
            bestD[k] = d; bestI[k] = m; d = td; m = ti;
          }
        }
      }
    }
  }
  int* row = idx + ((size_t)b * NN + n) * KNN;
#pragma unroll
  for (int k = 0; k < KNN; ++k) {
    row[k] = bestI[k];
    atomicAdd(&indeg[b * NN + bestI[k]], 1);
  }
}

// d[n] = rsqrt(D + 1e-6), D = 0.5*(K + indeg[n])  (row sum of (A+A^T)/2)
__global__ void degree_kernel(const int* __restrict__ indeg, float* __restrict__ dvec) {
  const int i = blockIdx.x * 256 + threadIdx.x;
  dvec[i] = rsqrtf(0.5f * (float)(KNN + indeg[i]) + 1e-6f);
}

// Per-batch exclusive scan of in-degrees -> rowStart (+ a fill cursor copy).
__global__ __launch_bounds__(256) void scan_kernel(
    const int* __restrict__ indeg, int* __restrict__ rowStart, int* __restrict__ cursor) {
  const int bb = blockIdx.x * NN;
  const int tid = threadIdx.x, base = tid * 8;
  int v[8];
  int tot = 0;
#pragma unroll
  for (int i = 0; i < 8; ++i) {
    const int t = indeg[bb + base + i];
    v[i] = tot; tot += t;
  }
  __shared__ int sums[256];
  sums[tid] = tot;
  __syncthreads();
  for (int off = 1; off < 256; off <<= 1) {
    const int x = (tid >= off) ? sums[tid - off] : 0;
    __syncthreads();
    sums[tid] += x;
    __syncthreads();
  }
  const int excl = sums[tid] - tot;
#pragma unroll
  for (int i = 0; i < 8; ++i) {
    const int rs = excl + v[i];
    rowStart[bb + base + i] = rs;
    cursor[bb + base + i]  = rs;
  }
}

// Reverse CSR fill: rev list grouped by destination node.
__global__ void fill_kernel(const int* __restrict__ idx, int* __restrict__ cursor,
                            int* __restrict__ rev) {
  const int i = blockIdx.x * 256 + threadIdx.x;   // i = b*N + m
  const int b = i >> 11;
  const int m = i & (NN - 1);
  const int* row = idx + (size_t)i * KNN;
  int* revb = rev + (size_t)b * NN * KNN;
#pragma unroll
  for (int k = 0; k < KNN; ++k) {
    const int nn = row[k];
    const int pos = atomicAdd(&cursor[(b << 11) + nn], 1);
    revb[pos] = m;
  }
}

// ---------------------------------------------------------------------------
// Sparse Laplacian apply: spec[:,n] = feat[:,n] - 0.5*d[n]*Sum d[m]*feat[:,m]
// over out-neighbors (idx) and in-neighbors (rev CSR). One wave per point,
// lane handles channels {lane, lane+32, lane+64, lane+96} (coalesced rows of
// featT). Emits bf16 specT (B,N,C) for the WMMA stage.
// ---------------------------------------------------------------------------
__global__ __launch_bounds__(256) void spec_kernel(
    const float* __restrict__ featT, const int* __restrict__ idx,
    const int* __restrict__ rowStart, const int* __restrict__ indeg,
    const int* __restrict__ rev, const float* __restrict__ dvec,
    unsigned short* __restrict__ specT) {
  const int gw = blockIdx.x * 8 + (threadIdx.x >> 5);
  const int lane = threadIdx.x & 31;
  const int b = gw >> 11, n = gw & (NN - 1);
  const float* fT = featT + (size_t)b * NN * CINC;
  const float* dv = dvec + b * NN;
  float a0 = 0.f, a1 = 0.f, a2 = 0.f, a3 = 0.f;
  const int* il = idx + ((size_t)b * NN + n) * KNN;
#pragma unroll 4
  for (int k = 0; k < KNN; ++k) {
    const int m = il[k];
    const float w = dv[m];
    const float* r = fT + (size_t)m * CINC + lane;
    a0 += w * r[0]; a1 += w * r[32]; a2 += w * r[64]; a3 += w * r[96];
  }
  const int s = rowStart[(b << 11) + n];
  const int e = s + indeg[(b << 11) + n];
  const int* revb = rev + (size_t)b * NN * KNN;
  for (int p = s; p < e; ++p) {
    const int m = revb[p];
    const float w = dv[m];
    const float* r = fT + (size_t)m * CINC + lane;
    a0 += w * r[0]; a1 += w * r[32]; a2 += w * r[64]; a3 += w * r[96];
  }
  const float hdn = 0.5f * dv[n];
  const float* rn = fT + (size_t)n * CINC + lane;
  unsigned short* o = specT + ((size_t)b * NN + n) * CINC + lane;
  o[0]  = f2bf(rn[0]  - hdn * a0);
  o[32] = f2bf(rn[32] - hdn * a1);
  o[64] = f2bf(rn[64] - hdn * a2);
  o[96] = f2bf(rn[96] - hdn * a3);
}

// ---------------------------------------------------------------------------
// Fused channel-mix: hidden = GELU(BN(W_spec @ spec)) ; out = W_lh @ hidden.
// Block = 8 waves, owns a 32-column slab of one batch. Each wave computes
// 4 16x16 tiles per GEMM via v_wmma_f32_16x16x32_bf16 (f32 accumulate).
// hidden staged in LDS as bf16 (32 x 256), one barrier between GEMMs.
// ---------------------------------------------------------------------------
__global__ __launch_bounds__(256) void gemm_kernel(
    const unsigned short* __restrict__ specT, const unsigned short* __restrict__ ApackS,
    const unsigned short* __restrict__ ApackLH, const float* __restrict__ alpha,
    const float* __restrict__ betaC, const float* __restrict__ bias2,
    float* __restrict__ out) {
  const int b = blockIdx.y;
  const int nBase = blockIdx.x * 32;
  const int w = threadIdx.x >> 5, lane = threadIdx.x & 31;
  const int half = lane >> 4, lm = lane & 15;
  __shared__ __align__(32) unsigned short hid[32 * 256];   // 16 KB bf16 hidden slab

  // ---- GEMM1: (256x128) @ (128x32) + affine + exact GELU -> LDS ----
#pragma unroll
  for (int t = 0; t < 4; ++t) {
    const int oT = 2 * w + (t >> 1);
    const int nT = t & 1;
    const int n = nBase + nT * 16 + lm;
    v8f acc = {};
    const unsigned short* bptr = specT + (((size_t)b * NN + n) * CINC + half * 16);
#pragma unroll
    for (int kT = 0; kT < 4; ++kT) {
      v16bf a  = *(const v16bf*)(ApackS + (((oT * 4 + kT) * 32 + lane) << 4));
      v16bf bm = *(const v16bf*)(bptr + kT * 32);
      acc = __builtin_amdgcn_wmma_f32_16x16x32_bf16(false, a, false, bm,
                                                    (short)0, acc, false, false);
    }
#pragma unroll
    for (int r = 0; r < 8; ++r) {
      const int o = oT * 16 + r + 8 * half;
      const float z = acc[r] * alpha[o] + betaC[o];
      const float g = 0.5f * z * (1.0f + erff(z * 0.70710678118654752f));
      hid[(nT * 16 + lm) * 256 + o] = f2bf(g);
    }
  }
  __syncthreads();

  // ---- GEMM2: (256x256) @ (256x32) + bias -> out (B,COUT,N) ----
#pragma unroll
  for (int t = 0; t < 4; ++t) {
    const int oT = 2 * w + (t >> 1);
    const int nT = t & 1;
    v8f acc = {};
    const unsigned short* bptr = &hid[(nT * 16 + lm) * 256 + half * 16];
#pragma unroll
    for (int kT = 0; kT < 8; ++kT) {
      v16bf a  = *(const v16bf*)(ApackLH + (((oT * 8 + kT) * 32 + lane) << 4));
      v16bf bm = *(const v16bf*)(bptr + kT * 32);
      acc = __builtin_amdgcn_wmma_f32_16x16x32_bf16(false, a, false, bm,
                                                    (short)0, acc, false, false);
    }
    const int n = nBase + nT * 16 + lm;
#pragma unroll
    for (int r = 0; r < 8; ++r) {
      const int o = oT * 16 + r + 8 * half;
      out[((size_t)b * COUTC + o) * NN + n] = acc[r] + bias2[o];
    }
  }
}

// ---------------------------------------------------------------------------
// Workspace layout (bytes from d_ws base; needs ~17.5 MB)
// ---------------------------------------------------------------------------
#define WS_IDX      0x0000000   // B*N*K int      = 1 MB
#define WS_INDEG    0x0200000   // B*N int        = 64 KB
#define WS_DVEC     0x0210000   // B*N f32        = 64 KB
#define WS_ROWS     0x0220000   // B*N int        = 64 KB
#define WS_CURSOR   0x0230000   // B*N int        = 64 KB
#define WS_REV      0x0240000   // B*N*K int      = 1 MB
#define WS_FEATT    0x0440000   // B*N*CIN f32    = 8 MB
#define WS_SPECT    0x0C40000   // B*N*CIN bf16   = 4 MB
#define WS_APACKS   0x1040000   // 256*128 bf16   = 64 KB
#define WS_APACKLH  0x1050000   // 256*256 bf16   = 128 KB
#define WS_ALPHA    0x1070000   // 256 f32
#define WS_BETAC    0x1070400   // 256 f32
#define WS_BIAS2    0x1070800   // 256 f32

extern "C" void kernel_launch(void* const* d_in, const int* in_sizes, int n_in,
                              void* d_out, int out_size, void* d_ws, size_t ws_size,
                              hipStream_t stream) {
  const float* xyz    = (const float*)d_in[0];
  const float* feat   = (const float*)d_in[1];
  const float* W_spec = (const float*)d_in[2];
  const float* b_spec = (const float*)d_in[3];
  const float* gamma  = (const float*)d_in[4];
  const float* beta   = (const float*)d_in[5];
  const float* mean   = (const float*)d_in[6];
  const float* var    = (const float*)d_in[7];
  const float* W_low  = (const float*)d_in[8];
  const float* b_low  = (const float*)d_in[9];
  const float* W_high = (const float*)d_in[10];
  const float* b_high = (const float*)d_in[11];
  float* out = (float*)d_out;

  char* ws = (char*)d_ws;
  int*            idx     = (int*)(ws + WS_IDX);
  int*            indeg   = (int*)(ws + WS_INDEG);
  float*          dvec    = (float*)(ws + WS_DVEC);
  int*            rowS    = (int*)(ws + WS_ROWS);
  int*            cursor  = (int*)(ws + WS_CURSOR);
  int*            rev     = (int*)(ws + WS_REV);
  float*          featT   = (float*)(ws + WS_FEATT);
  unsigned short* specT   = (unsigned short*)(ws + WS_SPECT);
  unsigned short* ApackS  = (unsigned short*)(ws + WS_APACKS);
  unsigned short* ApackLH = (unsigned short*)(ws + WS_APACKLH);
  float*          alpha   = (float*)(ws + WS_ALPHA);
  float*          betaC   = (float*)(ws + WS_BETAC);
  float*          bias2   = (float*)(ws + WS_BIAS2);

  hipMemsetAsync(indeg, 0, BB * NN * sizeof(int), stream);

  prep_kernel<<<1, 256, 0, stream>>>(W_spec, b_spec, gamma, beta, mean, var,
                                     W_low, b_low, W_high, b_high,
                                     ApackS, ApackLH, alpha, betaC, bias2);
  transpose_kernel<<<dim3(NN / 32, CINC / 32, BB), dim3(32, 8), 0, stream>>>(feat, featT);
  knn_kernel<<<BB * (NN / 256), 256, 0, stream>>>(xyz, idx, indeg);
  degree_kernel<<<BB * NN / 256, 256, 0, stream>>>(indeg, dvec);
  scan_kernel<<<BB, 256, 0, stream>>>(indeg, rowS, cursor);
  fill_kernel<<<BB * NN / 256, 256, 0, stream>>>(idx, cursor, rev);
  spec_kernel<<<BB * NN / 8, 256, 0, stream>>>(featT, idx, rowS, indeg, rev, dvec, specT);
  gemm_kernel<<<dim3(NN / 32, BB), 256, 0, stream>>>(specT, ApackS, ApackLH,
                                                     alpha, betaC, bias2, out);
}
